// SplitAttention_16501264351739
// MI455X (gfx1250) — compile-verified, matrix operand order checked by reading, not run on hardware
//
#include <hip/hip_runtime.h>
#include <hip/hip_bf16.h>

// ---------------- problem constants ----------------
#define TOK_DIM   512
#define POS_DIM   512
#define D_MODEL   1024
#define N_HEADS   16
#define HEAD_DIM  64
#define NUM_KV    4
#define INNER     1024
#define BATCH     2
#define SEQ       2048
#define MTOT      (BATCH*SEQ)      // 4096 rows
#define KPROJ     512              // reduction dim of q/k/v projections

typedef __bf16 bf16;
typedef __attribute__((ext_vector_type(16))) __bf16 v16bf;
typedef __attribute__((ext_vector_type(8)))  __bf16 v8bf;
typedef __attribute__((ext_vector_type(8)))  float  v8f;
typedef __attribute__((ext_vector_type(4)))  float  v4f;
typedef __attribute__((ext_vector_type(4)))  unsigned int u32x4;
typedef __attribute__((ext_vector_type(8)))  int i32x8;
typedef __attribute__((ext_vector_type(4)))  int i32x4;

#if defined(__has_builtin)
#if __has_builtin(__builtin_amdgcn_tensor_load_to_lds)
#define HAVE_TDM 1
#endif
#endif

// ---------------- bf16 helpers: native converts (v_cvt_pk_bf16_f32) -------
static __device__ __forceinline__ bf16  f2bf(float f) { return (bf16)f; }
static __device__ __forceinline__ float bf2f(bf16 h)  { return (float)h; }

static __device__ __forceinline__ v8f wmma_bf(v16bf a, v16bf b, v8f c) {
  // D(f32 16x16) = A(16x32 bf16) * B(32x16 bf16) + C
  return __builtin_amdgcn_wmma_f32_16x16x32_bf16(false, a, false, b, (short)0, c,
                                                 false, false);
}

// A/B fragment (16x32 bf16, wave32 layout) built from f32 memory (global/LDS).
// lane 0-15: row = lane, K-halves {0..7, 16..23}; lane 16-31: row = lane-16,
// K-halves {8..15, 24..31}.
static __device__ __forceinline__ v16bf frag_from_f32(const float* __restrict__ base,
                                                      int stride, int lane) {
  int r    = lane & 15;
  int koff = (lane >> 4) << 3;
  const float* p = base + (size_t)r * stride + koff;
  v4f a0 = *(const v4f*)(p);
  v4f a1 = *(const v4f*)(p + 4);
  v4f b0 = *(const v4f*)(p + 16);
  v4f b1 = *(const v4f*)(p + 20);
  v16bf o;
#pragma unroll
  for (int i = 0; i < 4; i++) {
    o[i]      = f2bf(a0[i]);
    o[4 + i]  = f2bf(a1[i]);
    o[8 + i]  = f2bf(b0[i]);
    o[12 + i] = f2bf(b1[i]);
  }
  return o;
}

// Fragment from bf16 memory (global or LDS); rows of `stride` elements.
static __device__ __forceinline__ v16bf frag_from_bf16(const bf16* __restrict__ base,
                                                       int stride, int lane) {
  int r    = lane & 15;
  int koff = (lane >> 4) << 3;
  const bf16* p = base + (size_t)r * stride + koff;
  v8bf h0 = *(const v8bf*)(p);
  v8bf h1 = *(const v8bf*)(p + 16);
  v16bf o;
#pragma unroll
  for (int i = 0; i < 8; i++) { o[i] = h0[i]; o[8 + i] = h1[i]; }
  return o;
}

#define LSTR 40  // bf16 LDS row stride (elems): 80B, 16B-aligned, bank-spread
#define WSTR 36  // f32  LDS row stride (elems): 144B, 16B-aligned, bank-spread

#ifdef HAVE_TDM
// TDM: DMA a 64-row x 32-col f32 tile (row stride = stride_elems) into LDS,
// injecting 4 DWORDs of padding every 32 DWORDs -> padded LDS stride WSTR=36.
static __device__ __forceinline__ void tdm_load_f32_tile(const float* gsrc,
                                                         void* ldsDst,
                                                         unsigned stride_elems) {
  unsigned long long ga = (unsigned long long)(size_t)gsrc;
  unsigned lds = (unsigned)(size_t)ldsDst;  // generic LDS ptr: low 32b = offset
  u32x4 g0;
  g0[0] = 1u;                                              // count=1 (valid D#)
  g0[1] = lds;                                             // lds_addr
  g0[2] = (unsigned)(ga & 0xffffffffull);                  // global_addr lo
  g0[3] = (unsigned)((ga >> 32) & 0x1ffffffull) | (2u << 30);  // hi | type=2
  i32x8 g1;
  g1[0] = (2 << 16)      // data_size = 4 bytes
        | (1 << 20)      // pad_enable
        | (4 << 22)      // pad_interval: every 32 DWORDs
        | (3 << 25);     // pad_amount: 4 DWORDs
  g1[1] = (int)(32u << 16);   // tensor_dim0 = 32 elems
  g1[2] = (int)(64u << 16);   // tensor_dim1 = 64 rows
  g1[3] = (int)(32u << 16);   // tile_dim0 = 32
  g1[4] = 64;                 // tile_dim1 = 64, tile_dim2 = 0
  g1[5] = (int)stride_elems;  // tensor_dim0_stride lo
  g1[6] = 0;
  g1[7] = 0;
  i32x4 gz = {0, 0, 0, 0};
#if __clang_major__ >= 23
  i32x8 gz8 = {0, 0, 0, 0, 0, 0, 0, 0};
  __builtin_amdgcn_tensor_load_to_lds(g0, g1, gz, gz, gz8, 0);
#else
  __builtin_amdgcn_tensor_load_to_lds(g0, g1, gz, gz, 0);
#endif
}
#endif  // HAVE_TDM

// Stage a 64x32 f32 weight tile into padded LDS (TDM if available).
static __device__ __forceinline__ void stage_w_tile(float* ldsW,
                                                    const float* __restrict__ W,
                                                    int colBase, int k, int KW,
                                                    int wave) {
#ifdef HAVE_TDM
  if (wave == 0) {
    tdm_load_f32_tile(W + (size_t)colBase * KW + k, ldsW, (unsigned)KW);
    __builtin_amdgcn_s_wait_tensorcnt(0);
  }
#else
  int n  = threadIdx.x >> 2;
  int kp = (threadIdx.x & 3) << 3;
  const float* src = W + (size_t)(colBase + n) * KW + k + kp;
  *(v4f*)(ldsW + n * WSTR + kp)     = *(const v4f*)(src);
  *(v4f*)(ldsW + n * WSTR + kp + 4) = *(const v4f*)(src + 4);
  (void)wave;
#endif
}

// ---------------- kernel 1: projection GEMM (x slice @ W^T -> bf16 heads) --
// Block: 256 threads = 8 waves, tile 256 rows x 64 cols; wave tile 32x64.
// out layout: [B, Hn, T, 64]  (head-major for attention)
__global__ void __launch_bounds__(256)
k_proj(const float* __restrict__ x, int col_off,
       const float* __restrict__ W, bf16* __restrict__ out, int Hn) {
  __shared__ float ldsW[64 * WSTR];
  const int lane = threadIdx.x & 31;
  const int wave = threadIdx.x >> 5;                 // 0..7
  const int rowBase = blockIdx.x * 256 + wave * 32;  // 8 waves * 32 rows
  const int colBase = blockIdx.y * 64;

  v8f acc[2][4] = {};
  for (int k = 0; k < KPROJ; k += 32) {
    stage_w_tile(ldsW, W, colBase, k, KPROJ, wave);
    __syncthreads();
    v16bf afr[2], bfr[4];
#pragma unroll
    for (int i = 0; i < 2; i++)
      afr[i] = frag_from_f32(x + (size_t)(rowBase + 16 * i) * D_MODEL + col_off + k,
                             D_MODEL, lane);
#pragma unroll
    for (int j = 0; j < 4; j++)
      bfr[j] = frag_from_f32(ldsW + j * 16 * WSTR, WSTR, lane);
#pragma unroll
    for (int i = 0; i < 2; i++)
#pragma unroll
      for (int j = 0; j < 4; j++)
        acc[i][j] = wmma_bf(afr[i], bfr[j], acc[i][j]);
    __syncthreads();
  }
  // store C: lane = column, vgpr index = row (+8 for upper half-wave)
  const int mofs = (lane >> 4) << 3;
#pragma unroll
  for (int i = 0; i < 2; i++) {
#pragma unroll
    for (int j = 0; j < 4; j++) {
      int n = colBase + j * 16 + (lane & 15);
      int h = n >> 6, d = n & 63;
#pragma unroll
      for (int r = 0; r < 8; r++) {
        int m = rowBase + 16 * i + r + mofs;
        int b = m >> 11, t = m & (SEQ - 1);
        out[(((size_t)b * Hn + h) * SEQ + t) * HEAD_DIM + d] = f2bf(acc[i][j][r]);
      }
    }
  }
}

// ---------------- kernel 2: per-head phase rotation on Q ------------------
__global__ void __launch_bounds__(256)
k_rope(const bf16* __restrict__ qin, const float* __restrict__ ang,
       bf16* __restrict__ qout) {
  size_t i = (size_t)blockIdx.x * blockDim.x + threadIdx.x;  // pair index
  int h = (int)((i >> 16) & 15);  // i = ((b*16+h)*2048 + t)*32 + p
  float c = __cosf(ang[h]);
  float s = __sinf(ang[h]);
  float q0 = bf2f(qin[2 * i]);
  float q1 = bf2f(qin[2 * i + 1]);
  qout[2 * i]     = f2bf(q0 * c - q1 * s);
  qout[2 * i + 1] = f2bf(q0 * s + q1 * c);
}

// ---------------- kernel 3: flash attention (causal + ALiBi) --------------
// grid: (B*H, T/64); block: 128 (4 waves x 16 query rows)
__global__ void __launch_bounds__(128)
k_attn(const bf16* __restrict__ q, const bf16* __restrict__ kk,
       const bf16* __restrict__ vv, const float* __restrict__ log_slopes,
       bf16* __restrict__ ctx) {
  __shared__ bf16 ldsP[4 * 16 * LSTR];   // per-wave P staging (C-layout->A-layout)
  __shared__ bf16 ldsVt[64 * LSTR];      // V^T tile: [d][key]

  const int lane = threadIdx.x & 31;
  const int wave = threadIdx.x >> 5;     // 0..3
  const int bh = blockIdx.x;
  const int b = bh >> 4, h = bh & 15;
  const int kvh = h >> 2;                // KV_REPEAT = 4
  const int q0 = blockIdx.y * 64;
  const int qrow = q0 + wave * 16;

  const bf16* qb = q  + (((size_t)b * N_HEADS + h)   * SEQ + qrow) * HEAD_DIM;
  const bf16* kb = kk + (((size_t)b * NUM_KV  + kvh) * SEQ) * HEAD_DIM;
  const bf16* vb = vv + (((size_t)b * NUM_KV  + kvh) * SEQ) * HEAD_DIM;
  const float slope = __expf(log_slopes[h]);

  const v16bf qf0 = frag_from_bf16(qb,      HEAD_DIM, lane);  // d 0..31
  const v16bf qf1 = frag_from_bf16(qb + 32, HEAD_DIM, lane);  // d 32..63

  v8f O[4] = {};
  float rmax[8], rsum[8];
#pragma unroll
  for (int r = 0; r < 8; r++) { rmax[r] = -3.0e38f; rsum[r] = 0.f; }
  const int mofs = (lane >> 4) << 3;
  const int ncol = lane & 15;

  const int ktiles = (q0 >> 5) + 2;  // keys 0 .. q0+63, tiles of 32
  for (int kt = 0; kt < ktiles; kt++) {
    const int key0 = kt * 32;
    __syncthreads();
    {  // stage V^T tile: ldsVt[d][key] = V[key0+key][d]
      int key = threadIdx.x & 31;
      int d0  = (threadIdx.x >> 5) << 4;
      const bf16* src = vb + (size_t)(key0 + key) * HEAD_DIM + d0;
#pragma unroll
      for (int i = 0; i < 16; i++) ldsVt[(d0 + i) * LSTR + key] = src[i];
    }
    __syncthreads();

    // S = Q @ K^T  (two 16-key halves, chained over d); batch the K-frag loads
    v16bf kf[2][2];
#pragma unroll
    for (int kn = 0; kn < 2; kn++) {
      const bf16* kbt = kb + (size_t)(key0 + kn * 16) * HEAD_DIM;
      kf[kn][0] = frag_from_bf16(kbt,      HEAD_DIM, lane);
      kf[kn][1] = frag_from_bf16(kbt + 32, HEAD_DIM, lane);
    }
    v8f s[2];
#pragma unroll
    for (int kn = 0; kn < 2; kn++) {
      v8f c = {};
      c = wmma_bf(qf0, kf[kn][0], c);
      c = wmma_bf(qf1, kf[kn][1], c);
      s[kn] = c;
    }

    // online softmax: scale + ALiBi + causal mask, row max/sum via shfl_xor
    float corr[8];
#pragma unroll
    for (int r = 0; r < 8; r++) {
      const int qi = qrow + r + mofs;
      float best = -3.0e38f;
#pragma unroll
      for (int kn = 0; kn < 2; kn++) {
        int ki = key0 + kn * 16 + ncol;
        float val = s[kn][r] * 0.125f + slope * (float)(qi - ki);
        if (ki > qi) val = -3.0e38f;
        s[kn][r] = val;
        best = fmaxf(best, val);
      }
#pragma unroll
      for (int m = 1; m < 16; m <<= 1) best = fmaxf(best, __shfl_xor(best, m, 32));
      float nmax = fmaxf(rmax[r], best);
      corr[r] = __expf(rmax[r] - nmax);
      rmax[r] = nmax;
      float ps = 0.f;
#pragma unroll
      for (int kn = 0; kn < 2; kn++) {
        float p = __expf(s[kn][r] - nmax);
        s[kn][r] = p;
        ps += p;
      }
#pragma unroll
      for (int m = 1; m < 16; m <<= 1) ps += __shfl_xor(ps, m, 32);
      rsum[r] = rsum[r] * corr[r] + ps;
    }
#pragma unroll
    for (int j = 0; j < 4; j++)
#pragma unroll
      for (int r = 0; r < 8; r++) O[j][r] *= corr[r];

    // P (C-layout) -> LDS row-major -> A-fragment
    bf16* preg = ldsP + wave * 16 * LSTR;
#pragma unroll
    for (int kn = 0; kn < 2; kn++)
#pragma unroll
      for (int r = 0; r < 8; r++)
        preg[(r + mofs) * LSTR + kn * 16 + ncol] = f2bf(s[kn][r]);
    v16bf pf = frag_from_bf16(preg, LSTR, lane);

    // O += P @ V   (V^T tile staged in LDS: column n = d, contiguous in key)
    v16bf vf[4];
#pragma unroll
    for (int jd = 0; jd < 4; jd++)
      vf[jd] = frag_from_bf16(ldsVt + jd * 16 * LSTR, LSTR, lane);
#pragma unroll
    for (int jd = 0; jd < 4; jd++)
      O[jd] = wmma_bf(pf, vf[jd], O[jd]);
  }

  // epilogue: normalize and store ctx[b][t][h*64+d]
#pragma unroll
  for (int jd = 0; jd < 4; jd++) {
#pragma unroll
    for (int r = 0; r < 8; r++) {
      int t = qrow + r + mofs;
      int d = jd * 16 + ncol;
      ctx[((size_t)(b * SEQ + t)) * INNER + h * HEAD_DIM + d] =
          f2bf(O[jd][r] * (1.0f / rsum[r]));
    }
  }
}

// ---------------- kernel 4: output projection (ctx bf16 @ Wo^T -> f32) ----
// Block: 256 threads = 8 waves, tile 256 rows x 64 cols; wave tile 32x64.
__global__ void __launch_bounds__(256)
k_oproj(const bf16* __restrict__ ctx, const float* __restrict__ Wo,
        float* __restrict__ out) {
  __shared__ float ldsW[64 * WSTR];
  const int lane = threadIdx.x & 31;
  const int wave = threadIdx.x >> 5;
  const int rowBase = blockIdx.x * 256 + wave * 32;
  const int colBase = blockIdx.y * 64;

  v8f acc[2][4] = {};
  for (int k = 0; k < INNER; k += 32) {
    stage_w_tile(ldsW, Wo, colBase, k, INNER, wave);
    __syncthreads();
    v16bf afr[2], bfr[4];
#pragma unroll
    for (int i = 0; i < 2; i++)
      afr[i] = frag_from_bf16(ctx + (size_t)(rowBase + 16 * i) * INNER + k,
                              INNER, lane);
#pragma unroll
    for (int j = 0; j < 4; j++)
      bfr[j] = frag_from_f32(ldsW + j * 16 * WSTR, WSTR, lane);
#pragma unroll
    for (int i = 0; i < 2; i++)
#pragma unroll
      for (int j = 0; j < 4; j++)
        acc[i][j] = wmma_bf(afr[i], bfr[j], acc[i][j]);
    __syncthreads();
  }
  const int mofs = (lane >> 4) << 3;
#pragma unroll
  for (int i = 0; i < 2; i++) {
#pragma unroll
    for (int j = 0; j < 4; j++) {
      int n = colBase + j * 16 + (lane & 15);
#pragma unroll
      for (int r = 0; r < 8; r++) {
        int m = rowBase + 16 * i + r + mofs;
        out[(size_t)m * D_MODEL + n] = acc[i][j][r];
      }
    }
  }
}

// ---------------- launcher -------------------------------------------------
extern "C" void kernel_launch(void* const* d_in, const int* in_sizes, int n_in,
                              void* d_out, int out_size, void* d_ws, size_t ws_size,
                              hipStream_t stream) {
  const float* x      = (const float*)d_in[0];
  const float* Wq     = (const float*)d_in[1];
  const float* Wk     = (const float*)d_in[2];
  const float* Wv     = (const float*)d_in[3];
  const float* Wo     = (const float*)d_in[4];
  const float* angles = (const float*)d_in[5];
  const float* lslope = (const float*)d_in[6];
  float* out = (float*)d_out;

  // workspace carve-up (bf16 buffers)
  char* ws = (char*)d_ws;
  const size_t SZ_Q = (size_t)BATCH * N_HEADS * SEQ * HEAD_DIM * sizeof(bf16);  // 8 MB
  const size_t SZ_K = (size_t)BATCH * NUM_KV  * SEQ * HEAD_DIM * sizeof(bf16);  // 2 MB
  bf16* q_raw = (bf16*)(ws);
  bf16* q_rot = (bf16*)(ws + SZ_Q);
  bf16* kbuf  = (bf16*)(ws + 2 * SZ_Q);
  bf16* vbuf  = (bf16*)(ws + 2 * SZ_Q + SZ_K);
  bf16* ctxb  = (bf16*)(ws + 2 * SZ_Q + 2 * SZ_K);
  (void)in_sizes; (void)n_in; (void)out_size; (void)ws_size;

  // 1) projections: q = x_pos Wq^T, k = x_pos Wk^T, v = x_tok Wv^T
  k_proj<<<dim3(MTOT / 256, N_HEADS), 256, 0, stream>>>(x, TOK_DIM, Wq, q_raw, N_HEADS);
  k_proj<<<dim3(MTOT / 256, NUM_KV),  256, 0, stream>>>(x, TOK_DIM, Wk, kbuf,  NUM_KV);
  k_proj<<<dim3(MTOT / 256, NUM_KV),  256, 0, stream>>>(x, 0,       Wv, vbuf,  NUM_KV);

  // 2) phase rotation on q
  {
    size_t pairs = (size_t)BATCH * N_HEADS * SEQ * (HEAD_DIM / 2);
    k_rope<<<(unsigned)(pairs / 256), 256, 0, stream>>>(q_raw, angles, q_rot);
  }

  // 3) flash attention
  k_attn<<<dim3(BATCH * N_HEADS, SEQ / 64), 128, 0, stream>>>(q_rot, kbuf, vbuf,
                                                              lslope, ctxb);

  // 4) output projection
  k_oproj<<<dim3(MTOT / 256, D_MODEL / 64), 256, 0, stream>>>(ctxb, Wo, out);
}
